// StableExpressivePENodeEncoder_90065464197250
// MI455X (gfx1250) — matrix operand order repaired and checked by reading, not compiled, not run on hardware
//
#include <hip/hip_runtime.h>

typedef __attribute__((ext_vector_type(16))) __bf16 v16bf;
typedef __attribute__((ext_vector_type(8)))  __bf16 v8bf;
typedef __attribute__((ext_vector_type(8)))  float  v8f;

#define NSUM 2048
#define NMAX 64
#define MDIM 16
#define HID  128
#define OUTD 64
#define NE   16384
#define NLAY 4
#define ROWS_TOT (NSUM * NMAX)   // 131072 rows

// ---------------- utility kernels ----------------

// Z = (1 + eps[layer]) * X   (also serves as the scatter accumulator init)
__global__ void k_zinit(const float* __restrict__ X, const float* __restrict__ eps,
                        int layer, float* __restrict__ Z, int nelem) {
  int i = blockIdx.x * 256 + threadIdx.x;
  if (i < nelem) Z[i] = (1.0f + eps[layer]) * X[i];
}

// scatter-add: Z[dst[e]] += X[src[e]] over [NMAX, D] node tiles
__global__ void k_scatter(const float* __restrict__ X, const int* __restrict__ src,
                          const int* __restrict__ dst, float* __restrict__ Z,
                          int nodeElems, int total) {
  int i = blockIdx.x * 256 + threadIdx.x;
  if (i >= total) return;
  int e   = i / nodeElems;
  int off = i - e * nodeElems;
  atomicAdd(&Z[(size_t)dst[e] * nodeElems + off], X[(size_t)src[e] * nodeElems + off]);
}

// f32 [rows, D] -> bf16 [rows, Keff] with zero pad for cols >= D
__global__ void k_cvt_pad(const float* __restrict__ Z, __bf16* __restrict__ Zb,
                          int D, int Keff, int total) {
  int i = blockIdx.x * 256 + threadIdx.x;
  if (i >= total) return;
  int row = i / Keff, col = i - row * Keff;
  float v = (col < D) ? Z[(size_t)row * D + col] : 0.0f;
  Zb[i] = (__bf16)v;
}

// Pack weights W[k, col] (f32, row-major) into per-lane WMMA B-fragment order:
// P[ks][n][lane][e] = W[ks*32 + (lane<16?0:8) + (e&7) + (e&8?16:0)][n*16 + (lane&15)], bf16,
// zero-padded for k >= Din. Each lane then loads its fragment as one v16bf.
__global__ void k_packw(const float* __restrict__ Wsrc, __bf16* __restrict__ P,
                        int Din, int Nout, int nt, int total) {
  int i = blockIdx.x * 256 + threadIdx.x;
  if (i >= total) return;
  int e    = i & 15;
  int lane = (i >> 4) & 31;
  int n    = (i >> 9) % nt;
  int ks   = (i >> 9) / nt;
  int k    = ks * 32 + ((lane < 16) ? 0 : 8) + (e & 7) + ((e & 8) ? 16 : 0);
  int col  = n * 16 + (lane & 15);
  float v  = (k < Din) ? Wsrc[(size_t)k * Nout + col] : 0.0f;
  P[i] = (__bf16)v;
}

// ---------------- WMMA GEMM ----------------
// C[rows, Nout] = A[rows, Keff](bf16) @ Wpacked + bias, f32 accumulate.
// Block = 128 threads (4 waves). Each wave: 32 rows x (NTILES*16) cols,
// processed in N-groups of <=4 tiles so each group's live set
// (64 acc + <=32 B + 16 A VGPRs) stays well under the spill threshold.
// Packed B (<=32KB) is staged in LDS once per block.
template <int NTILES, int KBLK>
__global__ __launch_bounds__(128, 1)
void k_gemm(const __bf16* __restrict__ A, const __bf16* __restrict__ P,
            const float* __restrict__ bias, float* __restrict__ C,
            int Keff, int Nout) {
  __shared__ __attribute__((aligned(32))) __bf16 sB[KBLK * NTILES * 32 * 16];

  // cooperative copy of packed weights into LDS (16B chunks)
  constexpr int totalVec = KBLK * NTILES * 512 / 8;  // v8bf chunks
  for (int i = threadIdx.x; i < totalVec; i += 128)
    ((v8bf*)sB)[i] = ((const v8bf*)P)[i];
  __syncthreads();

  int wave = threadIdx.x >> 5;
  int lane = threadIdx.x & 31;
  int row_base = blockIdx.x * 128 + wave * 32;       // 32 rows per wave
  int halo = (lane < 16) ? 0 : 8;  // 16-bit A layout: hi lanes hold K+8..15 / K+24..31
  int arow0 = row_base + (lane & 15);
  int arow1 = arow0 + 16;
  int r0 = row_base + ((lane < 16) ? 0 : 8);

  constexpr int TPG = (NTILES > 4) ? 4 : NTILES;  // tiles per group
  constexpr int NG  = NTILES / TPG;               // groups

#pragma unroll 1
  for (int g = 0; g < NG; ++g) {
    v8f acc0[TPG], acc1[TPG];
#pragma unroll
    for (int n = 0; n < TPG; ++n)
#pragma unroll
      for (int e = 0; e < 8; ++e) { acc0[n][e] = 0.0f; acc1[n][e] = 0.0f; }

#pragma unroll
    for (int ks = 0; ks < KBLK; ++ks) {
      int kbase = ks * 32 + halo;
      const __bf16* ap0 = A + (size_t)arow0 * Keff + kbase;
      const __bf16* ap1 = A + (size_t)arow1 * Keff + kbase;
      if (ks + 1 < KBLK) {                      // gfx1250 global_prefetch_b8
        __builtin_prefetch(ap0 + 32, 0, 0);
        __builtin_prefetch(ap1 + 32, 0, 0);
      }
      v8bf lo0 = *(const v8bf*)(ap0);           // K = kbase .. kbase+7
      v8bf hi0 = *(const v8bf*)(ap0 + 16);      // K = kbase+16 .. kbase+23
      v8bf lo1 = *(const v8bf*)(ap1);
      v8bf hi1 = *(const v8bf*)(ap1 + 16);
      v16bf a0, a1;
#pragma unroll
      for (int e = 0; e < 8; ++e) {
        a0[e] = lo0[e]; a0[e + 8] = hi0[e];
        a1[e] = lo1[e]; a1[e + 8] = hi1[e];
      }

#pragma unroll
      for (int n = 0; n < TPG; ++n) {
        int t = g * TPG + n;
        v16bf b = *(const v16bf*)(sB + ((ks * NTILES + t) * 32 + lane) * 16);
        acc0[n] = __builtin_amdgcn_wmma_f32_16x16x32_bf16(
            false, a0, false, b, (short)0, acc0[n], false, false);
        acc1[n] = __builtin_amdgcn_wmma_f32_16x16x32_bf16(
            false, a1, false, b, (short)0, acc1[n], false, false);
      }
    }

    // C/D layout: VGPR r -> M = r (lanes 0-15) or r+8 (lanes 16-31); N = lane&15.
#pragma unroll
    for (int n = 0; n < TPG; ++n) {
      int col = (g * TPG + n) * 16 + (lane & 15);
      float bv = bias[col];
#pragma unroll
      for (int r = 0; r < 8; ++r) {
        C[(size_t)(r0 + r) * Nout + col] = acc0[n][r] + bv;
        C[(size_t)(r0 + 16 + r) * Nout + col] = acc1[n][r] + bv;
      }
    }
  }
}

// ---------------- BatchNorm ----------------

__global__ void k_bnstats(const float* __restrict__ H, float* __restrict__ stats,
                          int rowsPerBlk) {
  int c = threadIdx.x;  // 128 threads, one channel each
  int row0 = blockIdx.x * rowsPerBlk;
  float s = 0.f, s2 = 0.f;
  for (int r = 0; r < rowsPerBlk; ++r) {
    float v = H[(size_t)(row0 + r) * HID + c];
    s += v; s2 += v * v;
  }
  atomicAdd(&stats[c], s);
  atomicAdd(&stats[HID + c], s2);
}

__global__ void k_bnapply(const float* __restrict__ H, const float* __restrict__ stats,
                          const float* __restrict__ gamma, const float* __restrict__ beta,
                          __bf16* __restrict__ Hb, int total) {
  int i = blockIdx.x * 256 + threadIdx.x;
  if (i >= total) return;
  int c = i & (HID - 1);
  const float invN = 1.0f / (float)ROWS_TOT;
  float mu  = stats[c] * invN;
  float var = stats[HID + c] * invN - mu * mu;
  float sc  = gamma[c] * rsqrtf(var + 1e-5f);
  float h   = (H[i] - mu) * sc + beta[c];
  h = h > 0.f ? h : 0.f;  // ReLU
  Hb[i] = (__bf16)h;
}

// ---------------- masked pooling ----------------

__global__ void k_pe(const float* __restrict__ X, const float* __restrict__ mask,
                     float* __restrict__ out) {
  int i = blockIdx.x * 256 + threadIdx.x;
  if (i >= NSUM * OUTD) return;
  int n = i >> 6, c = i & 63;
  float s = 0.f;
  for (int j = 0; j < NMAX; ++j)
    s += mask[n * NMAX + j] * X[(size_t)(n * NMAX + j) * OUTD + c];
  out[i] = s;
}

// ---------------- driver ----------------

extern "C" void kernel_launch(void* const* d_in, const int* in_sizes, int n_in,
                              void* d_out, int out_size, void* d_ws, size_t ws_size,
                              hipStream_t stream) {
  const float* Win  = (const float*)d_in[0];
  const float* mask = (const float*)d_in[1];
  const int*   src  = (const int*)d_in[2];
  const int*   dst  = (const int*)d_in[3];
  const float* eps  = (const float*)d_in[4];
  const float* W1f  = (const float*)d_in[5];
  const float* b1f  = (const float*)d_in[6];
  const float* W1r  = (const float*)d_in[7];
  const float* b1r  = (const float*)d_in[8];
  const float* gam  = (const float*)d_in[9];
  const float* bet  = (const float*)d_in[10];
  const float* W2m  = (const float*)d_in[11];
  const float* b2m  = (const float*)d_in[12];
  const float* W2l  = (const float*)d_in[13];
  const float* b2l  = (const float*)d_in[14];

  char* ws = (char*)d_ws;
  size_t o = 0;
  float*  bufF = (float*)(ws + o);  o += (size_t)ROWS_TOT * HID * 4;  // Z-f32 / H-f32 (aliased)
  __bf16* bufB = (__bf16*)(ws + o); o += (size_t)ROWS_TOT * HID * 2;  // Z-bf16 / H-bf16 (aliased)
  float*  Xf   = (float*)(ws + o);  o += (size_t)ROWS_TOT * HID * 4;  // activations between layers
  __bf16* P1   = (__bf16*)(ws + o); o += (size_t)4 * 8 * 32 * 16 * 2; // packed W1
  __bf16* P2   = (__bf16*)(ws + o); o += (size_t)4 * 8 * 32 * 16 * 2; // packed W2
  float*  stats = (float*)(ws + o); o += 2 * HID * 4;                 // BN sum / sumsq

  for (int l = 0; l < NLAY; ++l) {
    int D    = (l == 0) ? MDIM : HID;
    int Keff = (l == 0) ? 32 : HID;   // pad layer-0 K=16 up to one WMMA K-step
    int kblk = Keff / 32;
    const float* Xcur = (l == 0) ? Win : Xf;
    const float* W1 = (l == 0) ? W1f : W1r + (size_t)(l - 1) * HID * HID;
    const float* b1 = (l == 0) ? b1f : b1r + (size_t)(l - 1) * HID;
    int Nout2 = (l < NLAY - 1) ? HID : OUTD;
    const float* W2 = (l < NLAY - 1) ? W2m + (size_t)l * HID * HID : W2l;
    const float* b2 = (l < NLAY - 1) ? b2m + (size_t)l * HID : b2l;

    // pack weights into WMMA B-fragment order (bf16)
    int tot1 = kblk * 8 * 32 * 16;
    k_packw<<<(tot1 + 255) / 256, 256, 0, stream>>>(W1, P1, D, HID, 8, tot1);
    int nt2 = Nout2 / 16;
    int tot2 = (HID / 32) * nt2 * 32 * 16;
    k_packw<<<(tot2 + 255) / 256, 256, 0, stream>>>(W2, P2, HID, Nout2, nt2, tot2);

    // Z = (1+eps)X, then scatter-add messages
    int nelem = ROWS_TOT * D;
    k_zinit<<<(nelem + 255) / 256, 256, 0, stream>>>(Xcur, eps, l, bufF, nelem);
    int nodeElems = NMAX * D;
    int total = NE * nodeElems;
    k_scatter<<<(total + 255) / 256, 256, 0, stream>>>(Xcur, src, dst, bufF, nodeElems, total);

    // Z -> bf16 (zero-padded to Keff)
    int totc = ROWS_TOT * Keff;
    k_cvt_pad<<<(totc + 255) / 256, 256, 0, stream>>>(bufF, bufB, D, Keff, totc);

    // H = Z @ W1 + b1   (reads bufB, writes bufF)
    if (kblk == 1)
      k_gemm<8, 1><<<ROWS_TOT / 128, 128, 0, stream>>>(bufB, P1, b1, bufF, Keff, HID);
    else
      k_gemm<8, 4><<<ROWS_TOT / 128, 128, 0, stream>>>(bufB, P1, b1, bufF, Keff, HID);

    // BatchNorm + ReLU -> bf16 (reads bufF, writes bufB)
    hipMemsetAsync(stats, 0, 2 * HID * 4, stream);
    k_bnstats<<<ROWS_TOT / 512, HID, 0, stream>>>(bufF, stats, 512);
    int toth = ROWS_TOT * HID;
    k_bnapply<<<(toth + 255) / 256, 256, 0, stream>>>(bufF, stats, gam + l * HID,
                                                      bet + l * HID, bufB, toth);

    // X_next = H @ W2 + b2  (reads bufB, writes Xf)
    if (Nout2 == HID)
      k_gemm<8, 4><<<ROWS_TOT / 128, 128, 0, stream>>>(bufB, P2, b2, Xf, HID, HID);
    else
      k_gemm<4, 4><<<ROWS_TOT / 128, 128, 0, stream>>>(bufB, P2, b2, Xf, HID, OUTD);
  }

  // masked sum over padded node axis -> [NSUM, OUTD]
  k_pe<<<(NSUM * OUTD + 255) / 256, 256, 0, stream>>>(Xf, mask, (float*)d_out);
}
